// RealImagFourValLinear_17978733101539
// MI455X (gfx1250) — matrix operand order repaired and checked by reading, not compiled, not run on hardware
//
#include <hip/hip_runtime.h>

// ---------------------------------------------------------------------------
// RealImagFourValLinear for MI455X (gfx1250, wave32, WMMA)
//   y_re = (nr @ qw_re^T - ni @ qw_im^T) * mean|w_re|
//   y_im = (nr @ qw_im^T + ni @ qw_re^T) * mean|w_im|
// nr/ni = layernorm([x_re, x_im]) halves; qw = 4-value quantized weights
// (values {0, +-0.5, +-1} -> exact in bf16). GEMM: v_wmma_f32_16x16x32_bf16
// with double-buffered LDS fed by global_load_async_to_lds_b128 (ASYNCcnt).
// ---------------------------------------------------------------------------

#define HIGH_T 0.75f
#define LOW_T  0.25f
#define LN_EPS 1e-6f

typedef __attribute__((ext_vector_type(16))) __bf16       v16bf;
typedef __attribute__((ext_vector_type(8)))  float        v8f;
typedef __attribute__((ext_vector_type(4)))  unsigned int u32x4;

__device__ __forceinline__ unsigned short f32_to_bf16_rne(float f) {
  unsigned int u = __float_as_uint(f);
  u += 0x7FFFu + ((u >> 16) & 1u);          // round-to-nearest-even
  return (unsigned short)(u >> 16);
}

// Async 16B/lane copy global -> LDS, tracked by ASYNCcnt (CDNA5 §10 async ops).
// GVS addressing: mem = SGPR64 base + VGPR32 byte offset; dst = LDS byte addr.
__device__ __forceinline__ void async_b128(unsigned lds_addr, const void* gbase,
                                           unsigned byte_off) {
  asm volatile("global_load_async_to_lds_b128 %0, %1, %2"
               :
               : "v"(lds_addr), "v"(byte_off), "s"(gbase)
               : "memory");
}
__device__ __forceinline__ void wait_async0() {
  asm volatile("s_wait_asynccnt 0x0" ::: "memory");
}

// ---------------------------------------------------------------- init sums
__global__ void init_sums_kernel(float* sums) {
  sums[0] = 0.0f;
  sums[1] = 0.0f;
}

// ------------------------------------------------- weight quantize + |w| sum
__global__ __launch_bounds__(256) void quant_weight_kernel(
    const float* __restrict__ w, unsigned short* __restrict__ qw,
    float* __restrict__ sum_abs, int n) {
  float s = 0.0f;
  for (int i = blockIdx.x * blockDim.x + threadIdx.x; i < n;
       i += gridDim.x * blockDim.x) {
    float v = __builtin_nontemporal_load(w + i);   // single-use stream
    s += fabsf(v);
    float q = (v >  HIGH_T) ?  1.0f :
              (v >  LOW_T ) ?  0.5f :
              (v < -HIGH_T) ? -1.0f :
              (v < -LOW_T ) ? -0.5f : 0.0f;
    qw[i] = f32_to_bf16_rne(q);
  }
  __shared__ float red[256];
  red[threadIdx.x] = s;
  __syncthreads();
  for (int off = 128; off > 0; off >>= 1) {
    if (threadIdx.x < off) red[threadIdx.x] += red[threadIdx.x + off];
    __syncthreads();
  }
  if (threadIdx.x == 0) atomicAdd(sum_abs, red[0]);
}

// ---------------------------------------------------- layernorm -> bf16 nr/ni
__global__ __launch_bounds__(256) void layernorm_kernel(
    const float* __restrict__ xr, const float* __restrict__ xi,
    const float* __restrict__ gamma, const float* __restrict__ beta,
    unsigned short* __restrict__ nr, unsigned short* __restrict__ ni, int D) {
  const int row = blockIdx.x;
  const int tid = threadIdx.x;
  const float* pr = xr + (size_t)row * D;
  const float* pi = xi + (size_t)row * D;

  float s = 0.0f, ss = 0.0f;
  for (int k = tid; k < D; k += 256) {
    float a = pr[k]; s += a; ss += a * a;
    float b = pi[k]; s += b; ss += b * b;
  }
  __shared__ float rs[256], rq[256];
  rs[tid] = s; rq[tid] = ss;
  __syncthreads();
  for (int off = 128; off > 0; off >>= 1) {
    if (tid < off) { rs[tid] += rs[tid + off]; rq[tid] += rq[tid + off]; }
    __syncthreads();
  }
  const float invc = 1.0f / (float)(2 * D);
  const float mu   = rs[0] * invc;
  const float var  = rq[0] * invc - mu * mu;
  const float inv  = rsqrtf(var + LN_EPS);

  unsigned short* onr = nr + (size_t)row * D;
  unsigned short* oni = ni + (size_t)row * D;
  for (int k = tid; k < D; k += 256) {
    onr[k] = f32_to_bf16_rne((pr[k] - mu) * inv * gamma[k]     + beta[k]);
    oni[k] = f32_to_bf16_rne((pi[k] - mu) * inv * gamma[D + k] + beta[D + k]);
  }
}

// ------------------------------------------------------------- WMMA GEMM
// Block tile: BM=128 x BN=64, K-chunk KC=32 (one bf16 WMMA deep), double-
// buffered LDS (60 KB). 256 threads = 8 waves; wave grid 4(M) x 2(N): each
// wave owns a 32x32 output patch with 3 accumulator sets (P1,P2,P3).
#define BM 128
#define BN 64
#define KC 32
#define KS 40   // padded LDS row stride in bf16 halves (80 B, 16B-aligned)

// A-fragment (16x32 bf16, MxK): lanes 0-15 -> M=lane, K {0-7,16-23};
// lanes 16-31 -> M=lane-16, K {8-15,24-31}.  (ISA 7.12.2)
__device__ __forceinline__ v16bf lds_frag_a(const unsigned short* tile, int lane) {
  const int m = lane & 15;
  const int g = lane >> 4;
  const unsigned short* p = tile + m * KS + g * 8;
  v16bf f;
  ((u32x4*)&f)[0] = *(const u32x4*)(p);        // K chunk +0
  ((u32x4*)&f)[1] = *(const u32x4*)(p + 16);   // K chunk +16
  return f;
}

// B-fragment (32x16 bf16, KxN): lane holds column N=lane&15, K {0-7,16-23}
// for lanes 0-15, K {8-15,24-31} for lanes 16-31.  B = qw^T, qw stored [n][k] row-major.
__device__ __forceinline__ v16bf lds_frag_b(const unsigned short* tile, int lane) {
  const int n = lane & 15;
  const int g = lane >> 4;
  const unsigned short* p = tile + n * KS + g * 8;
  v16bf f;
  ((u32x4*)&f)[0] = *(const u32x4*)(p);
  ((u32x4*)&f)[1] = *(const u32x4*)(p + 16);
  return f;
}

__global__ __launch_bounds__(256) void cplx_q_gemm_kernel(
    const unsigned short* __restrict__ nr, const unsigned short* __restrict__ ni,
    const unsigned short* __restrict__ qr, const unsigned short* __restrict__ qi,
    const float* __restrict__ sums, float inv_wcount,
    float* __restrict__ y_re, float* __restrict__ y_im,
    int N, int K) {
  __shared__ unsigned short sAr[2][BM * KS];
  __shared__ unsigned short sAi[2][BM * KS];
  __shared__ unsigned short sBr[2][BN * KS];
  __shared__ unsigned short sBi[2][BN * KS];

  const int tid  = threadIdx.x;
  const int lane = tid & 31;
  const int wave = tid >> 5;
  const int wm   = wave & 3;   // 0..3 -> M offset wm*32
  const int wn   = wave >> 2;  // 0..1 -> N offset wn*32
  const int m0   = blockIdx.y * BM;
  const int n0   = blockIdx.x * BN;

  // LDS byte addresses of each stage (flat-pointer low 32 bits == LDS offset).
  const unsigned ldsAr[2] = {(unsigned)(uintptr_t)&sAr[0][0], (unsigned)(uintptr_t)&sAr[1][0]};
  const unsigned ldsAi[2] = {(unsigned)(uintptr_t)&sAi[0][0], (unsigned)(uintptr_t)&sAi[1][0]};
  const unsigned ldsBr[2] = {(unsigned)(uintptr_t)&sBr[0][0], (unsigned)(uintptr_t)&sBr[1][0]};
  const unsigned ldsBi[2] = {(unsigned)(uintptr_t)&sBi[0][0], (unsigned)(uintptr_t)&sBi[1][0]};

  const int cc      = tid & 3;           // 16B chunk within a 64B row
  const int aRow0   = tid >> 2;          // A rows 0..63  (chunk tid)
  const int aRow1   = aRow0 + 64;        // A rows 64..127 (chunk tid+256)
  const int bRow    = tid >> 2;          // B rows 0..63
  const unsigned ldsOffA0 = (unsigned)(aRow0 * KS + cc * 8) * 2u;
  const unsigned ldsOffA1 = (unsigned)(aRow1 * KS + cc * 8) * 2u;
  const unsigned ldsOffB  = (unsigned)(bRow  * KS + cc * 8) * 2u;

  // Issue one stage of async global->LDS copies (6 x b128 per thread).
  auto issue_stage = [&](int stage, int k0) {
    const unsigned goA0 = ((unsigned)(m0 + aRow0) * (unsigned)K + (unsigned)k0 + cc * 8u) * 2u;
    const unsigned goA1 = ((unsigned)(m0 + aRow1) * (unsigned)K + (unsigned)k0 + cc * 8u) * 2u;
    const unsigned goB  = ((unsigned)(n0 + bRow ) * (unsigned)K + (unsigned)k0 + cc * 8u) * 2u;
    async_b128(ldsAr[stage] + ldsOffA0, nr, goA0);
    async_b128(ldsAi[stage] + ldsOffA0, ni, goA0);
    async_b128(ldsAr[stage] + ldsOffA1, nr, goA1);
    async_b128(ldsAi[stage] + ldsOffA1, ni, goA1);
    async_b128(ldsBr[stage] + ldsOffB,  qr, goB);
    async_b128(ldsBi[stage] + ldsOffB,  qi, goB);
  };

  v8f accP1[2][2], accP2[2][2], accP3[2][2];   // nr*re, ni*im, (nr*im + ni*re)
#pragma unroll
  for (int i = 0; i < 2; ++i)
#pragma unroll
    for (int j = 0; j < 2; ++j)
#pragma unroll
      for (int e = 0; e < 8; ++e) {
        accP1[i][j][e] = 0.0f; accP2[i][j][e] = 0.0f; accP3[i][j][e] = 0.0f;
      }

  issue_stage(0, 0);
  int cur = 0;
  for (int k0 = 0; k0 < K; k0 += KC) {
    wait_async0();       // drain this wave's copies for stage `cur`
    __syncthreads();     // all waves' copies done; prior stage reads retired

    v16bf ar[2], ai[2], br[2], bi[2];
#pragma unroll
    for (int mi = 0; mi < 2; ++mi) {
      const int ro = (wm * 32 + mi * 16) * KS;
      ar[mi] = lds_frag_a(&sAr[cur][0] + ro, lane);
      ai[mi] = lds_frag_a(&sAi[cur][0] + ro, lane);
    }
#pragma unroll
    for (int nj = 0; nj < 2; ++nj) {
      const int ro = (wn * 32 + nj * 16) * KS;
      br[nj] = lds_frag_b(&sBr[cur][0] + ro, lane);
      bi[nj] = lds_frag_b(&sBi[cur][0] + ro, lane);
    }

    if (k0 + KC < K) issue_stage(cur ^ 1, k0 + KC);   // prefetch next stage

#pragma unroll
    for (int mi = 0; mi < 2; ++mi)
#pragma unroll
      for (int nj = 0; nj < 2; ++nj) {
        accP1[mi][nj] = __builtin_amdgcn_wmma_f32_16x16x32_bf16(
            false, ar[mi], false, br[nj], (short)0, accP1[mi][nj], false, false);
        accP2[mi][nj] = __builtin_amdgcn_wmma_f32_16x16x32_bf16(
            false, ai[mi], false, bi[nj], (short)0, accP2[mi][nj], false, false);
        accP3[mi][nj] = __builtin_amdgcn_wmma_f32_16x16x32_bf16(
            false, ar[mi], false, bi[nj], (short)0, accP3[mi][nj], false, false);
        accP3[mi][nj] = __builtin_amdgcn_wmma_f32_16x16x32_bf16(
            false, ai[mi], false, br[nj], (short)0, accP3[mi][nj], false, false);
      }
    cur ^= 1;
  }

  const float s_re = sums[0] * inv_wcount;
  const float s_im = sums[1] * inv_wcount;

  // C/D layout (ISA 7.12.2): lane col N = lane&15; VGPR r -> row r + (lane>>4)*8
#pragma unroll
  for (int mi = 0; mi < 2; ++mi) {
    const int rowBase = m0 + wm * 32 + mi * 16 + ((lane >> 4) * 8);
#pragma unroll
    for (int nj = 0; nj < 2; ++nj) {
      const int col = n0 + wn * 32 + nj * 16 + (lane & 15);
#pragma unroll
      for (int r = 0; r < 8; ++r) {
        const size_t o = (size_t)(rowBase + r) * N + col;
        __builtin_nontemporal_store((accP1[mi][nj][r] - accP2[mi][nj][r]) * s_re,
                                    y_re + o);
        __builtin_nontemporal_store(accP3[mi][nj][r] * s_im, y_im + o);
      }
    }
  }
}

// ---------------------------------------------------------------------------
extern "C" void kernel_launch(void* const* d_in, const int* in_sizes, int n_in,
                              void* d_out, int out_size, void* d_ws, size_t ws_size,
                              hipStream_t stream) {
  (void)n_in; (void)out_size; (void)ws_size;
  const float* x_real = (const float*)d_in[0];
  const float* x_imag = (const float*)d_in[1];
  const float* w_re   = (const float*)d_in[2];
  const float* w_im   = (const float*)d_in[3];
  const float* gamma  = (const float*)d_in[4];
  const float* beta   = (const float*)d_in[5];

  const int DIN   = in_sizes[4] / 2;        // gamma is (2*D_IN,)
  const int DOUT  = in_sizes[2] / DIN;      // weight_re is (D_OUT, D_IN)
  const int Mrows = in_sizes[0] / DIN;      // x_real is (B, S, D_IN)

  // Workspace layout (needs ~80 MB):
  //  [0,256)                      float sums[2] = { sum|w_re|, sum|w_im| }
  //  qw_re, qw_im : DOUT*DIN bf16 each
  //  nr, ni       : Mrows*DIN bf16 each
  char* ws = (char*)d_ws;
  float* sums = (float*)ws;
  unsigned short* qw_re = (unsigned short*)(ws + 256);
  unsigned short* qw_im = qw_re + (size_t)DOUT * DIN;
  unsigned short* nr    = qw_im + (size_t)DOUT * DIN;
  unsigned short* ni    = nr + (size_t)Mrows * DIN;

  float* y_re = (float*)d_out;
  float* y_im = y_re + (size_t)Mrows * DOUT;

  init_sums_kernel<<<1, 1, 0, stream>>>(sums);
  quant_weight_kernel<<<2048, 256, 0, stream>>>(w_re, qw_re, sums + 0, DOUT * DIN);
  quant_weight_kernel<<<2048, 256, 0, stream>>>(w_im, qw_im, sums + 1, DOUT * DIN);
  layernorm_kernel<<<Mrows, 256, 0, stream>>>(x_real, x_imag, gamma, beta, nr, ni, DIN);

  const float inv_wcount = 1.0f / ((float)DOUT * (float)DIN);
  dim3 grid(DOUT / BN, Mrows / BM);
  cplx_q_gemm_kernel<<<grid, 256, 0, stream>>>(nr, ni, qw_re, qw_im, sums,
                                               inv_wcount, y_re, y_im, DOUT, DIN);
}
